// MultiHeadAttentionCrossWithWeights_38671885534056
// MI455X (gfx1250) — matrix operand order repaired and checked by reading, not compiled, not run on hardware
//
#include <hip/hip_runtime.h>
#include <hip/hip_bf16.h>
#include <math.h>

typedef __bf16 bf16_t;
typedef __attribute__((ext_vector_type(8)))  bf16_t bfx8;
typedef __attribute__((ext_vector_type(16))) bf16_t bfx16;
typedef __attribute__((ext_vector_type(8)))  float  fx8;

#define WMMA_BF16(A,B,C) __builtin_amdgcn_wmma_f32_16x16x32_bf16(false,(A),false,(B),(short)0,(C),false,false)

static constexpr int Bsz = 8;
static constexpr int TQ  = 448;
static constexpr int TK  = 1500;
static constexpr int TKP = 1536;   // padded key count (24 tiles of 64)
static constexpr int DIM = 1280;
static constexpr int NH  = 20;
static constexpr int DH  = 64;

// A-fragment loader: lane holds one row; two contiguous 16-byte chunks at +0 and +8 elems.
__device__ inline bfx16 load_a16(const bf16_t* base) {
    bfx8 c0 = *(const bfx8*)(base);
    bfx8 c1 = *(const bfx8*)(base + 8);
    return __builtin_shufflevector(c0, c1, 0,1,2,3,4,5,6,7,8,9,10,11,12,13,14,15);
}

// CDNA5 async global -> LDS copy, 16 bytes per lane (tracked by ASYNCcnt).
// dst operand = per-lane LDS byte address VGPR, src = per-lane 64-bit global address.
__device__ inline void async_load_b128(unsigned lds_off, const void* gptr) {
    asm volatile("global_load_async_to_lds_b128 %0, %1, off"
                 :: "v"(lds_off), "v"((unsigned long long)(size_t)gptr)
                 : "memory");
}
__device__ inline void wait_async0() {
#if __has_builtin(__builtin_amdgcn_s_wait_asynccnt)
    __builtin_amdgcn_s_wait_asynccnt(0);
#else
    asm volatile("s_wait_asynccnt 0" ::: "memory");
#endif
}

// ---------------- conversion kernels ----------------
__global__ void cvt_f32_to_bf16(const float* __restrict__ in, bf16_t* __restrict__ out, size_t n) {
    size_t i = (size_t)blockIdx.x * blockDim.x + threadIdx.x;
    size_t stride = (size_t)gridDim.x * blockDim.x;
    for (; i < n; i += stride) out[i] = (bf16_t)in[i];
}

// k/v: (B, TK, D) f32 -> (B, TKP, D) bf16, zero-padded rows TK..TKP-1
__global__ void cvt_kv_pad(const float* __restrict__ in, bf16_t* __restrict__ out) {
    size_t total = (size_t)Bsz * TKP * DIM;
    size_t i = (size_t)blockIdx.x * blockDim.x + threadIdx.x;
    size_t stride = (size_t)gridDim.x * blockDim.x;
    for (; i < total; i += stride) {
        size_t c = i % DIM;
        size_t r = (i / DIM) % TKP;
        size_t b = i / ((size_t)DIM * TKP);
        out[i] = (r < (size_t)TK) ? (bf16_t)in[(b * TK + r) * DIM + c] : (bf16_t)0.f;
    }
}

// ---------------- WMMA GEMM: C[M,N] = A[M,K] * Brows[N,K]^T + bias ----------------
// mode 0: outb = bf16((acc + bias) * scale)   (Q projection, scale = Dh^-0.5)
// mode 1: outf = acc + bias                   (output projection, fp32)
__global__ __launch_bounds__(256)
void gemm_wmma_bf16(const bf16_t* __restrict__ A, const bf16_t* __restrict__ Bw,
                    const float* __restrict__ bias, bf16_t* __restrict__ outb,
                    float* __restrict__ outf, int M, int N, int K, int mode, float scale) {
    const int wave  = threadIdx.x >> 5;
    const int lane  = threadIdx.x & 31;
    const int lmod  = lane & 15;
    const int lhalf = lane >> 4;
    const int m0 = (blockIdx.x * 8 + wave) * 16;
    const int nb = blockIdx.y * 64;

    fx8 z = {0.f,0.f,0.f,0.f,0.f,0.f,0.f,0.f};
    fx8 acc[4] = {z, z, z, z};

    const bf16_t* arow = A + (size_t)(m0 + lmod) * K + lhalf * 16;
    for (int k0 = 0; k0 < K; k0 += 32) {
        bfx16 a = load_a16(arow + k0);
        #pragma unroll
        for (int nt = 0; nt < 4; ++nt) {
            const bf16_t* bp = Bw + (size_t)(nb + nt * 16 + lmod) * K + k0 + lhalf * 16;
            bfx16 b = *(const bfx16*)bp;
            acc[nt] = WMMA_BF16(a, b, acc[nt]);
        }
    }
    #pragma unroll
    for (int nt = 0; nt < 4; ++nt) {
        int n = nb + nt * 16 + lmod;
        float bv = bias[n];
        #pragma unroll
        for (int r = 0; r < 8; ++r) {
            int m = m0 + r + 8 * lhalf;
            float val = acc[nt][r] + bv;
            if (mode == 0) outb[(size_t)m * N + n] = (bf16_t)(val * scale);
            else           outf[(size_t)m * N + n] = val;
        }
    }
}

// ---------------- fused attention per (qblock, head, batch) ----------------
__global__ __launch_bounds__(128)
void attn_wmma(const bf16_t* __restrict__ qm, const bf16_t* __restrict__ km,
               const bf16_t* __restrict__ vm, bf16_t* __restrict__ ao,
               float* __restrict__ wout) {
    __shared__ bf16_t kT[64 * 64];   // K tile, natural row-major: kT[n][d] (async-staged)
    __shared__ bf16_t vT[64 * 80];   // V tile transposed: vT[d][n]
    __shared__ bf16_t pl[64 * 80];   // P tile row-major:  pl[m][n]

    const int qblk = blockIdx.x, h = blockIdx.y, b = blockIdx.z;
    const int q0 = qblk * 64;
    const int tid = threadIdx.x;
    const int wave = tid >> 5, lane = tid & 31;
    const int lmod = lane & 15, lhalf = lane >> 4;

    // Q A-fragments for this wave's 16-row strip (cached for both passes)
    bfx16 aq[2];
    {
        const bf16_t* base = qm + (size_t)(b * TQ + q0 + wave * 16 + lmod) * DIM
                             + h * DH + lhalf * 16;
        aq[0] = load_a16(base);
        aq[1] = load_a16(base + 32);
    }

    float mrow[8], lrow[8];
    #pragma unroll
    for (int r = 0; r < 8; ++r) { mrow[r] = -__builtin_inff(); lrow[r] = 0.f; }

    const int NT = TKP / 64;  // 24 key tiles

    // ---- pass 1: row max + sum-exp (online) ----
    for (int kt = 0; kt < NT; ++kt) {
        __syncthreads();     // protect kT reuse
        // async-stage K tile: 512 chunks of 16B, 4 per thread
        #pragma unroll
        for (int i = 0; i < 4; ++i) {
            int c = tid + i * 128;
            int n = c >> 3, dc = (c & 7) * 8;
            const bf16_t* g = km + (size_t)(b * TKP + kt * 64 + n) * DIM + h * DH + dc;
            async_load_b128((unsigned)(size_t)&kT[n * 64 + dc], g);
        }
        wait_async0();
        __syncthreads();

        fx8 s[4];
        #pragma unroll
        for (int nt = 0; nt < 4; ++nt) {
            fx8 z = {0.f,0.f,0.f,0.f,0.f,0.f,0.f,0.f};
            s[nt] = z;
            const bf16_t* kp = &kT[(nt * 16 + lmod) * 64 + lhalf * 16];
            bfx16 b0 = *(const bfx16*)kp;          // d0 = 0
            bfx16 b1 = *(const bfx16*)(kp + 32);   // d0 = 32
            s[nt] = WMMA_BF16(aq[0], b0, s[nt]);
            s[nt] = WMMA_BF16(aq[1], b1, s[nt]);
        }
        #pragma unroll
        for (int nt = 0; nt < 4; ++nt) {
            int kk = kt * 64 + nt * 16 + lmod;
            if (kk >= TK) {
                #pragma unroll
                for (int r = 0; r < 8; ++r) s[nt][r] = -__builtin_inff();
            }
        }
        #pragma unroll
        for (int r = 0; r < 8; ++r) {
            float tm = fmaxf(fmaxf(s[0][r], s[1][r]), fmaxf(s[2][r], s[3][r]));
            #pragma unroll
            for (int off = 1; off < 16; off <<= 1) tm = fmaxf(tm, __shfl_xor(tm, off, 32));
            float nm = fmaxf(mrow[r], tm);
            float ts = __expf(s[0][r] - nm) + __expf(s[1][r] - nm)
                     + __expf(s[2][r] - nm) + __expf(s[3][r] - nm);
            #pragma unroll
            for (int off = 1; off < 16; off <<= 1) ts += __shfl_xor(ts, off, 32);
            lrow[r] = lrow[r] * __expf(mrow[r] - nm) + ts;
            mrow[r] = nm;
        }
    }
    float inv_l[8];
    #pragma unroll
    for (int r = 0; r < 8; ++r) inv_l[r] = 1.f / lrow[r];

    const int ai = (h == 4) ? 0 : (h == 7) ? 1 : (h == 11) ? 2 : -1;
    float* wbase = (ai >= 0) ? (wout + (size_t)(b * 3 + ai) * TQ * TK) : nullptr;

    fx8 zv = {0.f,0.f,0.f,0.f,0.f,0.f,0.f,0.f};
    fx8 o[4] = {zv, zv, zv, zv};

    // ---- pass 2: p = exp(s-m)/l, emit alignment weights, O += P*V ----
    for (int kt = 0; kt < NT; ++kt) {
        __syncthreads();     // protect kT/vT reuse
        // async-stage K tile again
        #pragma unroll
        for (int i = 0; i < 4; ++i) {
            int c = tid + i * 128;
            int n = c >> 3, dc = (c & 7) * 8;
            const bf16_t* g = km + (size_t)(b * TKP + kt * 64 + n) * DIM + h * DH + dc;
            async_load_b128((unsigned)(size_t)&kT[n * 64 + dc], g);
        }
        // prefetch next V tile while staging this one
        if (kt + 1 < NT)
            __builtin_prefetch(vm + (size_t)(b * TKP + (kt + 1) * 64) * DIM + h * DH, 0, 0);
        // cooperative transposed V tile load: vT[d][n] = V[kt*64+n][h*64+d]
        for (int e = tid; e < 4096; e += 128) {
            int n = e >> 6, d = e & 63;
            vT[d * 80 + n] = vm[(size_t)(b * TKP + kt * 64 + n) * DIM + h * DH + d];
        }
        wait_async0();
        __syncthreads();

        fx8 s[4];
        #pragma unroll
        for (int nt = 0; nt < 4; ++nt) {
            fx8 z = {0.f,0.f,0.f,0.f,0.f,0.f,0.f,0.f};
            s[nt] = z;
            const bf16_t* kp = &kT[(nt * 16 + lmod) * 64 + lhalf * 16];
            bfx16 b0 = *(const bfx16*)kp;
            bfx16 b1 = *(const bfx16*)(kp + 32);
            s[nt] = WMMA_BF16(aq[0], b0, s[nt]);
            s[nt] = WMMA_BF16(aq[1], b1, s[nt]);
        }
        #pragma unroll
        for (int nt = 0; nt < 4; ++nt) {
            int kk = kt * 64 + nt * 16 + lmod;
            bool valid = kk < TK;
            #pragma unroll
            for (int r = 0; r < 8; ++r) {
                float p = valid ? __expf(s[nt][r] - mrow[r]) * inv_l[r] : 0.f;
                int row = wave * 16 + r + 8 * lhalf;           // C/D layout row
                pl[row * 80 + nt * 16 + lmod] = (bf16_t)p;
                if (ai >= 0 && valid) {
                    wbase[(size_t)(q0 + row) * TK + kk] = p;   // fp32 alignment weight
                }
            }
        }
        __syncthreads();

        #pragma unroll
        for (int na = 0; na < 2; ++na) {
            const bf16_t* pb = pl + (wave * 16 + lmod) * 80 + na * 32 + lhalf * 16;
            bfx16 ap = load_a16(pb);
            #pragma unroll
            for (int dt = 0; dt < 4; ++dt) {
                const bf16_t* vp = vT + (dt * 16 + lmod) * 80 + na * 32 + lhalf * 16;
                bfx16 bv2 = *(const bfx16*)vp;
                o[dt] = WMMA_BF16(ap, bv2, o[dt]);
            }
        }
    }

    // write attention output (bf16, feeds the Wo GEMM)
    #pragma unroll
    for (int dt = 0; dt < 4; ++dt) {
        #pragma unroll
        for (int r = 0; r < 8; ++r) {
            int row = q0 + wave * 16 + r + 8 * lhalf;
            int col = h * DH + dt * 16 + lmod;
            ao[(size_t)(b * TQ + row) * DIM + col] = (bf16_t)o[dt][r];
        }
    }
}

// ---------------- launcher ----------------
extern "C" void kernel_launch(void* const* d_in, const int* in_sizes, int n_in,
                              void* d_out, int out_size, void* d_ws, size_t ws_size,
                              hipStream_t stream) {
    (void)in_sizes; (void)n_in; (void)out_size; (void)ws_size;
    const float* x  = (const float*)d_in[0];
    const float* k  = (const float*)d_in[1];
    const float* v  = (const float*)d_in[2];
    const float* Wq = (const float*)d_in[3];
    const float* bq = (const float*)d_in[4];
    const float* Wo = (const float*)d_in[5];
    const float* bo = (const float*)d_in[6];
    float* outp = (float*)d_out;

    const size_t NX  = (size_t)Bsz * TQ * DIM;    // 4,587,520
    const size_t NW  = (size_t)DIM * DIM;         // 1,638,400
    const size_t NKV = (size_t)Bsz * TKP * DIM;   // 15,728,640

    bf16_t* xb  = (bf16_t*)d_ws;
    bf16_t* wqb = xb  + NX;
    bf16_t* wob = wqb + NW;
    bf16_t* qb  = wob + NW;
    bf16_t* kb  = qb  + NX;
    bf16_t* vb  = kb  + NKV;
    bf16_t* ab  = vb  + NKV;

    float* align_out = outp + NX;   // alignment weights after `out`

    // conversions
    cvt_f32_to_bf16<<<2048, 256, 0, stream>>>(x,  xb,  NX);
    cvt_f32_to_bf16<<<2048, 256, 0, stream>>>(Wq, wqb, NW);
    cvt_f32_to_bf16<<<2048, 256, 0, stream>>>(Wo, wob, NW);
    cvt_kv_pad     <<<2048, 256, 0, stream>>>(k, kb);
    cvt_kv_pad     <<<2048, 256, 0, stream>>>(v, vb);

    // Q projection: q = bf16((x @ Wq^T + bq) * Dh^-0.5)   (full QK scale folded in)
    const int M = Bsz * TQ;  // 3584
    dim3 ggrid(M / 128, DIM / 64);
    gemm_wmma_bf16<<<ggrid, 256, 0, stream>>>(xb, wqb, bq, qb, nullptr,
                                              M, DIM, DIM, 0, 0.125f);

    // fused attention
    dim3 agrid(TQ / 64, NH, Bsz);
    attn_wmma<<<agrid, 128, 0, stream>>>(qb, kb, vb, ab, align_out);

    // output projection: out = attn @ Wo^T + bo  (fp32)
    gemm_wmma_bf16<<<ggrid, 256, 0, stream>>>(ab, wob, bo, nullptr, outp,
                                              M, DIM, DIM, 1, 0.f);
}
